// DeformableFusion_46308337386267
// MI455X (gfx1250) — matrix operand order repaired and checked by reading, not compiled, not run on hardware
//
#include <hip/hip_runtime.h>
#include <math.h>

typedef __attribute__((ext_vector_type(16))) _Float16 v16h;
typedef __attribute__((ext_vector_type(8)))  _Float16 v8h;
typedef __attribute__((ext_vector_type(8)))  float    v8f;

#define HH 128
#define WW 128
#define NB 4
#define NF 64
#define HW (HH*WW)

__device__ __forceinline__ _Float16 f2h(float f) { return (_Float16)f; }

// ---------------------------------------------------------------------------
// f32 -> f16 weight conversion (run once; outputs live in ws arena)
// ---------------------------------------------------------------------------
__global__ void cvt_f16(const float* __restrict__ s, _Float16* __restrict__ d, int n) {
  int i = blockIdx.x * blockDim.x + threadIdx.x;
  if (i < n) d[i] = (_Float16)s[i];
}

// PyTorch bicubic kernel, a = -0.75
__device__ __forceinline__ float cubicw(float t) {
  float at = fabsf(t);
  const float a = -0.75f;
  if (at <= 1.f) return ((a + 2.f) * at - (a + 3.f)) * at * at + 1.f;
  if (at < 2.f)  return (((at - 5.f) * at + 8.f) * at - 4.f) * a;
  return 0.f;
}

// ---------------------------------------------------------------------------
// ref = T[i] * bicubic_resize(S[i], 128x128)  (border-clamped taps)
// ---------------------------------------------------------------------------
__global__ void prep_ref(const float* __restrict__ T, const float* __restrict__ S,
                         float* __restrict__ ref) {
  int id = blockIdx.x * blockDim.x + threadIdx.x;
  if (id >= NB * HW) return;
  int x = id % WW, y = (id / WW) % HH, b = id / HW;

  float sy = (y + 0.5f) * 0.5f - 0.5f;
  float sx = (x + 0.5f) * 0.5f - 0.5f;
  int iy0 = (int)floorf(sy), ix0 = (int)floorf(sx);
  float ty = sy - (float)iy0, tx = sx - (float)ix0;
  float wy[4], wx[4]; int yy[4], xx[4];
#pragma unroll
  for (int k = 0; k < 4; ++k) {
    wy[k] = cubicw(ty - (float)(k - 1));
    wx[k] = cubicw(tx - (float)(k - 1));
    yy[k] = min(max(iy0 + k - 1, 0), 63);
    xx[k] = min(max(ix0 + k - 1, 0), 63);
  }
  const float* Sp = S + (size_t)b * (64 * 64);
  float sval = 0.f;
#pragma unroll
  for (int ky = 0; ky < 4; ++ky) {
    float r = 0.f;
#pragma unroll
    for (int kx = 0; kx < 4; ++kx) r += wx[kx] * Sp[yy[ky] * 64 + xx[kx]];
    sval += wy[ky] * r;
  }
  const float* Tp = T + (size_t)b * NF * HW + y * WW + x;
  float*       Rp = ref + (size_t)b * NF * HW + y * WW + x;
  for (int c = 0; c < NF; ++c) Rp[c * HW] = Tp[c * HW] * sval;
}

// ---------------------------------------------------------------------------
// Generic 3x3 SAME conv: implicit GEMM on v_wmma_f32_16x16x32_f16.
// Tile: 16 contiguous x-pixels x Cout; wave = one 16-channel N-tile.
// LDS: compact patch (Cin x 3 x 18) -> im2col panel panel[m][k] (linear k!)
// so A-fragments are two ds_load_b128 with immediate offsets, zero VALU.
// ACT: 0 none | 1 relu | 2 offset/mod (clip ch<18, 2*sigmoid rest)
// ---------------------------------------------------------------------------
template<int CIN, int ACT, bool HAS_RESID>
__global__ __launch_bounds__(128) void conv3x3_wmma(
    const float* __restrict__ in0, const float* __restrict__ in1,
    const _Float16* __restrict__ w, int Cout,
    const float* __restrict__ bias, const float* __restrict__ resid,
    float* __restrict__ out, float clipv)
{
  constexpr int C0 = (CIN < NF) ? CIN : NF;
  constexpr int KT = CIN * 9;
  constexpr int PP = KT + 8;                 // pad: 16B-aligned rows, no bank clash
  __shared__ _Float16 patch[CIN * 54];
  __shared__ _Float16 panel[16 * PP];
  const int tid = threadIdx.x, nthr = blockDim.x;
  const int lane = tid & 31, wave = tid >> 5;
  const int gid = blockIdx.x;
  const int xt = (gid & 7) * 16;
  const int row = gid >> 3;
  const int y = row % HH, b = row / HH;

  // ---- phase A: compact zero-padded patch, f32 -> f16 ----
  for (int idx = tid; idx < CIN * 54; idx += nthr) {
    int c = idx / 54, rem = idx - c * 54;
    int ry = rem / 18, rx = rem - ry * 18;
    int gy = y - 1 + ry, gx = xt - 1 + rx;
    float v = 0.f;
    if (gy >= 0 && gy < HH && gx >= 0 && gx < WW) {
      if (CIN > NF && c >= C0)
        v = in1[(size_t)(b * (CIN - C0) + (c - C0)) * HW + gy * WW + gx];
      else
        v = in0[(size_t)(b * C0 + c) * HW + gy * WW + gx];
    }
    patch[idx] = f2h(v);
  }
  __syncthreads();

  // ---- phase B: im2col panel, panel[m][k], k = c*9 + ry*3 + rx (r unrolled) ----
  {
    int m = tid & 15;
    for (int c = tid >> 4; c < CIN; c += (nthr >> 4)) {
#pragma unroll
      for (int r = 0; r < 9; ++r)
        panel[m * PP + c * 9 + r] = patch[c * 54 + (r / 3) * 18 + m + (r % 3)];
    }
  }
  __syncthreads();

  // ---- GEMM: [16 x KT] x [KT x 16] per wave ----
  const int m = lane & 15, hf = lane >> 4;
  const int o = wave * 16 + (lane & 15);
  const _Float16* Arow = panel + m * PP + hf * 16;           // 16B aligned
  const _Float16* Wrow = w + (size_t)o * KT + hf * 16;       // 16B aligned

  v8f acc = {};
#pragma unroll
  for (int kb = 0; kb < KT; kb += 32) {
    v8h a0 = *(const v8h*)(Arow + kb);        // k = kb + hf*16 + 0..7
    v8h a1 = *(const v8h*)(Arow + kb + 8);    // k = kb + hf*16 + 8..15
    v8h b0 = *(const v8h*)(Wrow + kb);        // k = kb + hf*16 + 0..7
    v8h b1 = *(const v8h*)(Wrow + kb + 8);    // k = kb + hf*16 + 8..15
    v16h a  = __builtin_shufflevector(a0, a1, 0,1,2,3,4,5,6,7,8,9,10,11,12,13,14,15);
    v16h bm = __builtin_shufflevector(b0, b1, 0,1,2,3,4,5,6,7,8,9,10,11,12,13,14,15);
    acc = __builtin_amdgcn_wmma_f32_16x16x32_f16(false, a, false, bm,
                                                 (short)0, acc, false, false);
  }

  if (o < Cout) {
    float bv = bias[o];
    size_t base = (size_t)(b * Cout + o) * HW + y * WW + xt + hf * 8;
    float r[8];
#pragma unroll
    for (int vi = 0; vi < 8; ++vi) {
      float val = acc[vi] + bv;
      if (ACT == 1) val = fmaxf(val, 0.f);
      else if (ACT == 2)
        val = (o < 18) ? fminf(fmaxf(val, -clipv), clipv)
                       : 2.f / (1.f + __expf(-val));
      r[vi] = val;
    }
    if (HAS_RESID) {
      float4 q0 = *(const float4*)(resid + base);
      float4 q1 = *(const float4*)(resid + base + 4);
      r[0] += q0.x; r[1] += q0.y; r[2] += q0.z; r[3] += q0.w;
      r[4] += q1.x; r[5] += q1.y; r[6] += q1.z; r[7] += q1.w;
    }
    *(float4*)(out + base)     = make_float4(r[0], r[1], r[2], r[3]);
    *(float4*)(out + base + 4) = make_float4(r[4], r[5], r[6], r[7]);
  }
}

// ---------------------------------------------------------------------------
// Deformable conv: cooperative bilinear gather builds LDS panel A[16][576]
// (modulator folded into taps), then the same chunked WMMA GEMM (N=64),
// accumulating into the attention buffer.
// ---------------------------------------------------------------------------
__global__ __launch_bounds__(128) void deform_wmma(
    const float* __restrict__ ref, const float* __restrict__ offmod,
    const _Float16* __restrict__ w, const float* __restrict__ bias,
    float* __restrict__ att, int firstI)
{
  constexpr int KT = 576, PP = 584;
  __shared__ _Float16 A[16 * PP];
  __shared__ int   s_iy[144], s_ix[144];
  __shared__ float s_w00[144], s_w01[144], s_w10[144], s_w11[144];
  const int tid = threadIdx.x;
  const int gid = blockIdx.x;
  const int xt = (gid & 7) * 16;
  const int row = gid >> 3;
  const int y = row % HH, b = row / HH;

  // ---- phase 1: sampling params for 16 pixels x 9 taps ----
  for (int idx = tid; idx < 144; idx += 128) {
    int m = idx / 9, kk = idx - m * 9;
    int x = xt + m;
    const float* omp = offmod + (size_t)b * 27 * HW + y * WW + x;
    float dy = omp[(2 * kk) * HW];
    float dx = omp[(2 * kk + 1) * HW];
    float mv = omp[(18 + kk) * HW];
    float py = (float)(y - 1 + kk / 3) + dy;
    float px = (float)(x - 1 + kk % 3) + dx;
    float fy = floorf(py), fx = floorf(px);
    float ty = py - fy, tx = px - fx;
    s_iy[idx] = (int)fy; s_ix[idx] = (int)fx;
    s_w00[idx] = (1.f - ty) * (1.f - tx) * mv;
    s_w01[idx] = (1.f - ty) * tx * mv;
    s_w10[idx] = ty * (1.f - tx) * mv;
    s_w11[idx] = ty * tx * mv;
  }
  __syncthreads();

  // ---- phase 2: gather into panel (thread = pixel m, 8 channels) ----
  {
    int m = tid & 15, cbase = (tid >> 4) * 8;
    for (int kk = 0; kk < 9; ++kk) {
      int idx = m * 9 + kk;
      int iy = s_iy[idx], ix = s_ix[idx];
      float w00 = s_w00[idx], w01 = s_w01[idx], w10 = s_w10[idx], w11 = s_w11[idx];
      bool v00 = (iy >= 0 && iy < HH && ix >= 0 && ix < WW);
      bool v01 = (iy >= 0 && iy < HH && ix + 1 >= 0 && ix + 1 < WW);
      bool v10 = (iy + 1 >= 0 && iy + 1 < HH && ix >= 0 && ix < WW);
      bool v11 = (iy + 1 >= 0 && iy + 1 < HH && ix + 1 >= 0 && ix + 1 < WW);
      int y0 = min(max(iy, 0), HH - 1), y1 = min(max(iy + 1, 0), HH - 1);
      int x0 = min(max(ix, 0), WW - 1), x1 = min(max(ix + 1, 0), WW - 1);
      for (int cc = 0; cc < 8; ++cc) {
        int c = cbase + cc;
        const float* rp = ref + (size_t)(b * NF + c) * HW;
        float v = 0.f;
        if (v00) v += w00 * rp[y0 * WW + x0];
        if (v01) v += w01 * rp[y0 * WW + x1];
        if (v10) v += w10 * rp[y1 * WW + x0];
        if (v11) v += w11 * rp[y1 * WW + x1];
        A[m * PP + c * 9 + kk] = f2h(v);     // k = c*9 + kk matches weight layout
      }
    }
  }
  __syncthreads();

  // ---- phase 3: GEMM [16 x 576] x [576 x 64] via chunked WMMA ----
  const int lane = tid & 31, wave = tid >> 5;
  const int m = lane & 15, hf = lane >> 4;
  const int o = wave * 16 + (lane & 15);
  const _Float16* Arow = A + m * PP + hf * 16;
  const _Float16* Wrow = w + (size_t)o * KT + hf * 16;

  v8f acc = {};
#pragma unroll
  for (int kb = 0; kb < KT; kb += 32) {
    v8h a0 = *(const v8h*)(Arow + kb);
    v8h a1 = *(const v8h*)(Arow + kb + 8);
    v8h b0 = *(const v8h*)(Wrow + kb);
    v8h b1 = *(const v8h*)(Wrow + kb + 8);
    v16h a  = __builtin_shufflevector(a0, a1, 0,1,2,3,4,5,6,7,8,9,10,11,12,13,14,15);
    v16h bm = __builtin_shufflevector(b0, b1, 0,1,2,3,4,5,6,7,8,9,10,11,12,13,14,15);
    acc = __builtin_amdgcn_wmma_f32_16x16x32_f16(false, a, false, bm,
                                                 (short)0, acc, false, false);
  }

  float bv = bias[o];
  size_t base = (size_t)(b * NF + o) * HW + y * WW + xt + hf * 8;
  float r[8];
#pragma unroll
  for (int vi = 0; vi < 8; ++vi) r[vi] = acc[vi] + bv;
  if (!firstI) {
    float4 q0 = *(const float4*)(att + base);
    float4 q1 = *(const float4*)(att + base + 4);
    r[0] += q0.x; r[1] += q0.y; r[2] += q0.z; r[3] += q0.w;
    r[4] += q1.x; r[5] += q1.y; r[6] += q1.z; r[7] += q1.w;
  }
  *(float4*)(att + base)     = make_float4(r[0], r[1], r[2], r[3]);
  *(float4*)(att + base + 4) = make_float4(r[4], r[5], r[6], r[7]);
}

// ---------------------------------------------------------------------------
extern "C" void kernel_launch(void* const* d_in, const int* in_sizes, int n_in,
                              void* d_out, int out_size, void* d_ws, size_t ws_size,
                              hipStream_t stream) {
  (void)in_sizes; (void)n_in; (void)out_size; (void)ws_size;
  const float* x      = (const float*)d_in[0];
  const float* T      = (const float*)d_in[1];
  const float* S      = (const float*)d_in[2];
  const float* off_w  = (const float*)d_in[3];
  const float* off_b  = (const float*)d_in[4];
  const float* mod_w  = (const float*)d_in[5];
  const float* mod_b  = (const float*)d_in[6];
  const float* reg_w  = (const float*)d_in[7];
  const float* reg_b  = (const float*)d_in[8];
  const float* fuse_w = (const float*)d_in[9];
  const float* fuse_b = (const float*)d_in[10];
  const float* rb_w1  = (const float*)d_in[11];
  const float* rb_b1  = (const float*)d_in[12];
  const float* rb_w2  = (const float*)d_in[13];
  const float* rb_b2  = (const float*)d_in[14];
  float* outp = (float*)d_out;

  float* ws = (float*)d_ws;
  const size_t NIMG = (size_t)NB * NF * HW;           // 4,194,304 floats
  float* ref    = ws;
  float* offmod = ref + NIMG;                         // 4*27*HW
  float* att    = offmod + (size_t)NB * 27 * HW;
  float* xbuf   = att + NIMG;
  float* tmp    = xbuf + NIMG;
  float* bpk    = tmp + NIMG;                         // packed offmod biases (3*27)
  _Float16* wh  = (_Float16*)(bpk + 128);             // f16 weight arena

  const int K128 = 2 * NF * 9;                        // 1152
  const int K64  = NF * 9;                            // 576
  _Float16* wOFF  = wh;                               // 3 * 27 * 1152
  _Float16* wREG  = wOFF  + (size_t)3 * 27 * K128;    // 3 * 64 * 576
  _Float16* wFUSE = wREG  + (size_t)3 * NF * K64;     // 64 * 1152
  _Float16* wRB1  = wFUSE + (size_t)NF * K128;        // 8 * 64 * 576
  _Float16* wRB2  = wRB1  + (size_t)8 * NF * K64;

  // ---- one-time weight conversion + bias packing ----
  const int CT = 256;
  for (int i = 0; i < 3; ++i) {
    cvt_f16<<<(18 * K128 + CT - 1) / CT, CT, 0, stream>>>(
        off_w + (size_t)i * 18 * K128, wOFF + (size_t)i * 27 * K128, 18 * K128);
    cvt_f16<<<(9 * K128 + CT - 1) / CT, CT, 0, stream>>>(
        mod_w + (size_t)i * 9 * K128, wOFF + (size_t)i * 27 * K128 + 18 * K128, 9 * K128);
    hipMemcpyAsync(bpk + i * 27,      off_b + i * 18, 18 * sizeof(float),
                   hipMemcpyDeviceToDevice, stream);
    hipMemcpyAsync(bpk + i * 27 + 18, mod_b + i * 9,   9 * sizeof(float),
                   hipMemcpyDeviceToDevice, stream);
  }
  cvt_f16<<<(3 * NF * K64 + CT - 1) / CT, CT, 0, stream>>>(reg_w, wREG, 3 * NF * K64);
  cvt_f16<<<(NF * K128 + CT - 1) / CT, CT, 0, stream>>>(fuse_w, wFUSE, NF * K128);
  cvt_f16<<<(8 * NF * K64 + CT - 1) / CT, CT, 0, stream>>>(rb_w1, wRB1, 8 * NF * K64);
  cvt_f16<<<(8 * NF * K64 + CT - 1) / CT, CT, 0, stream>>>(rb_w2, wRB2, 8 * NF * K64);

  const int tiles = NB * HH * WW / 16;                // 4096 pixel tiles

  for (int i = 0; i < 3; ++i) {
    prep_ref<<<(NB * HW + 255) / 256, 256, 0, stream>>>(
        T + (size_t)i * NB * NF * HW, S + (size_t)i * NB * 64 * 64, ref);
    conv3x3_wmma<128, 2, false><<<tiles, 64, 0, stream>>>(
        x, ref, wOFF + (size_t)i * 27 * K128, 27, bpk + i * 27,
        nullptr, offmod, 32.0f /* max(h,w)/4 */);
    deform_wmma<<<tiles, 128, 0, stream>>>(
        ref, offmod, wREG + (size_t)i * NF * K64, reg_b + i * NF, att,
        (i == 0) ? 1 : 0);
  }

  // x = x + conv3x3(concat(x, attention))
  conv3x3_wmma<128, 0, true><<<tiles, 128, 0, stream>>>(
      x, att, wFUSE, NF, fuse_b, x, xbuf, 0.f);

  // 8 residual blocks; conv2 writes x in place (reads only tmp + own pixel)
  for (int j = 0; j < 8; ++j) {
    conv3x3_wmma<64, 1, false><<<tiles, 128, 0, stream>>>(
        xbuf, nullptr, wRB1 + (size_t)j * NF * K64, NF, rb_b1 + j * NF,
        nullptr, tmp, 0.f);
    float* dst = (j == 7) ? outp : xbuf;
    conv3x3_wmma<64, 0, true><<<tiles, 128, 0, stream>>>(
        tmp, nullptr, wRB2 + (size_t)j * NF * K64, NF, rb_b2 + j * NF,
        xbuf, dst, 0.f);
  }
}
